// CrossAttention_69114613730483
// MI455X (gfx1250) — matrix operand order repaired and checked by reading, not compiled, not run on hardware
//
#include <hip/hip_runtime.h>
#include <hip/hip_bf16.h>

// Problem constants (from reference): B=8, C=256, W=H=64 -> N=4096, CQ=32
#define BB 8
#define CC 256
#define NN 4096
#define DD 32
#define SPAD 68   // S-tile LDS row stride (floats): 16B-aligned rows, 8*68%64=32 -> halves hit disjoint banks

typedef __bf16 bf16_t;
typedef __bf16 v16bf __attribute__((ext_vector_type(16)));
typedef __bf16 v8bf  __attribute__((ext_vector_type(8)));
typedef float  v8f   __attribute__((ext_vector_type(8)));

// ---------- fragment helpers (wave32, 16x16x32 bf16 WMMA) ----------
// A-matrix (16x32, MxK): lane L: m = L&15, hi = L>>4.
//   halfs 0..15 -> K = 16*hi + 0..15   (contiguous 32 bytes)
// B-matrix (32x16, KxN): lane L: n = L&15, hi = L>>4.
//   halfs 0..15 -> K = 16*hi + 0..15   (contiguous 32 bytes)

static __device__ __forceinline__ v16bf ld2_bf16(const bf16_t* p0, const bf16_t* p1) {
  v8bf a = *(const v8bf*)p0;
  v8bf b = *(const v8bf*)p1;
  return __builtin_shufflevector(a, b, 0,1,2,3,4,5,6,7,8,9,10,11,12,13,14,15);
}

static __device__ __forceinline__ v16bf ld2_f32cvt(const float* p0, const float* p1) {
  v16bf r;
#pragma unroll
  for (int i = 0; i < 8; ++i) r[i] = (bf16_t)p0[i];
#pragma unroll
  for (int i = 0; i < 8; ++i) r[8 + i] = (bf16_t)p1[i];
  return r;
}

static __device__ __forceinline__ v8f zero_v8f() {
  v8f z;
#pragma unroll
  for (int i = 0; i < 8; ++i) z[i] = 0.0f;
  return z;
}

static __device__ __forceinline__ v8f wmma_bf16(v16bf a, v16bf b, v8f c) {
  // (neg_a, A, neg_b, B, c_mod, C, reuse_a, reuse_b)
  return __builtin_amdgcn_wmma_f32_16x16x32_bf16(false, a, false, b, (short)0, c,
                                                 false, false);
}

// ---------- kernel 0: transpose + f32->bf16 convert: [B][C][N] -> [B][N][C] ----------
__global__ void __launch_bounds__(256)
k_transpose_cvt(const float* __restrict__ src, bf16_t* __restrict__ dst) {
  __shared__ float tile[32][33];
  const int b  = blockIdx.z;
  const int i0 = blockIdx.x * 32;
  const int c0 = blockIdx.y * 32;
  const int tx = threadIdx.x;      // 0..31
  const int ty = threadIdx.y;      // 0..7
  const float* s = src + ((size_t)b * CC + c0) * NN + i0;
#pragma unroll
  for (int k = 0; k < 4; ++k)
    tile[ty + 8 * k][tx] = s[(size_t)(ty + 8 * k) * NN + tx];
  __syncthreads();
  bf16_t* d = dst + ((size_t)b * NN + i0) * CC + c0;
#pragma unroll
  for (int k = 0; k < 4; ++k)
    d[(size_t)(ty + 8 * k) * CC + tx] = (bf16_t)tile[tx][ty + 8 * k];
}

// ---------- kernel 1: Q/K projection: Out[b][i][d] = sum_c xT[b][i][c]*Wp[d][c] + bp[d] ----------
__global__ void __launch_bounds__(256)
k_qk_proj(const bf16_t* __restrict__ xT, const float* __restrict__ Wp,
          const float* __restrict__ bp, bf16_t* __restrict__ Out) {
  const int b    = blockIdx.y;
  const int w    = threadIdx.x >> 5;
  const int lane = threadIdx.x & 31;
  const int n    = lane & 15;
  const int hi   = lane >> 4;
  const int i0   = blockIdx.x * 128 + w * 16;

  const bf16_t* xrow = xT + ((size_t)b * NN + i0 + n) * CC;  // A: row m=n, K=c
  v8f acc[2] = {zero_v8f(), zero_v8f()};

#pragma unroll 4
  for (int kk = 0; kk < CC; kk += 32) {
    v16bf a = ld2_bf16(xrow + kk + 16 * hi, xrow + kk + 16 * hi + 8);
#pragma unroll
    for (int ci = 0; ci < 2; ++ci) {
      const float* wrow = Wp + (size_t)(ci * 16 + n) * CC + kk + 16 * hi;  // B: col n=d
      v16bf bm = ld2_f32cvt(wrow, wrow + 8);
      acc[ci] = wmma_bf16(a, bm, acc[ci]);
    }
  }
#pragma unroll
  for (int ci = 0; ci < 2; ++ci) {
    const float bias = bp[ci * 16 + n];
#pragma unroll
    for (int r = 0; r < 8; ++r) {
      const int row = i0 + r + 8 * hi;
      Out[(size_t)b * NN * DD + (size_t)row * DD + ci * 16 + n] =
          (bf16_t)(acc[ci][r] + bias);
    }
  }
}

// ---------- kernel 2: H projection: H[b][c][i] = sum_c' Wh[c][c']*xT[b][i][c'] + bh[c] ----------
__global__ void __launch_bounds__(256)
k_h_proj(const bf16_t* __restrict__ xT, const float* __restrict__ Whm,
         const float* __restrict__ bh, bf16_t* __restrict__ H) {
  const int b    = blockIdx.z;
  const int w    = threadIdx.x >> 5;
  const int lane = threadIdx.x & 31;
  const int n    = lane & 15;
  const int hi   = lane >> 4;
  const int c0   = blockIdx.y * 128 + w * 16;
  const int i0   = blockIdx.x * 64;

  v8f acc[4] = {zero_v8f(), zero_v8f(), zero_v8f(), zero_v8f()};
  const float* wrow = Whm + (size_t)(c0 + n) * CC;  // A: row m=n -> c

#pragma unroll 2
  for (int kk = 0; kk < CC; kk += 32) {
    v16bf a = ld2_f32cvt(wrow + kk + 16 * hi, wrow + kk + 16 * hi + 8);
#pragma unroll
    for (int iv = 0; iv < 4; ++iv) {
      const bf16_t* xrow =
          xT + ((size_t)b * NN + i0 + iv * 16 + n) * CC + kk + 16 * hi;  // B: col n=i
      v16bf bm = ld2_bf16(xrow, xrow + 8);
      acc[iv] = wmma_bf16(a, bm, acc[iv]);
    }
  }
  // bias rows are consecutive in r: two float4 loads instead of 32 scalar loads
  const float4 bh0 = *(const float4*)(bh + c0 + 8 * hi);
  const float4 bh1 = *(const float4*)(bh + c0 + 8 * hi + 4);
  const float bias[8] = {bh0.x, bh0.y, bh0.z, bh0.w, bh1.x, bh1.y, bh1.z, bh1.w};
#pragma unroll
  for (int iv = 0; iv < 4; ++iv) {
#pragma unroll
    for (int r = 0; r < 8; ++r) {
      const int c = c0 + r + 8 * hi;
      H[(size_t)b * CC * NN + (size_t)c * NN + i0 + iv * 16 + n] =
          (bf16_t)(acc[iv][r] + bias[r]);
    }
  }
}

// ---------- kernel 3: fused flash attention + gamma*sa + x ----------
// Block = 256 threads (8 waves), handles (batch b, 64-query tile i0).
// Wave w owns output columns c in [w*32, w*32+32); accumulators 64x32 f32 (8 C-tiles).
__global__ void __launch_bounds__(256)
k_attention(const bf16_t* __restrict__ Qb, const bf16_t* __restrict__ Kb,
            const bf16_t* __restrict__ Hb, const float* __restrict__ x,
            const float* __restrict__ gamma_p, float* __restrict__ out) {
  __shared__ float  Sld[64][SPAD];  // raw logits tile
  __shared__ bf16_t Pld[64][72];    // exp(S - m) bf16 (row stride 144B, 16B aligned)
  __shared__ float  rs[64];         // per-row accumulator rescale exp(m_old - m_new)
  __shared__ float  ls[64];         // final per-row softmax denominators

  const int b    = blockIdx.y;
  const int i0   = blockIdx.x * 64;
  const int tid  = threadIdx.x;
  const int w    = tid >> 5;
  const int lane = tid & 31;
  const int n    = lane & 15;
  const int hi   = lane >> 4;
  const int ti   = w >> 1;  // query row-strip (0..3) this wave computes S for

  // Q A-fragment for this wave's S row strip (resident for whole kernel)
  const bf16_t* qrow = Qb + ((size_t)b * NN + i0 + 16 * ti + n) * DD;
  const v16bf aq = ld2_bf16(qrow + 16 * hi, qrow + 16 * hi + 8);

  v8f acc[4][2];
#pragma unroll
  for (int ri = 0; ri < 4; ++ri) {
    acc[ri][0] = zero_v8f();
    acc[ri][1] = zero_v8f();
  }
  float m_run = -3.0e38f, l_run = 0.0f;  // meaningful for tid < 64 (one row each)

  const bf16_t* Kbase = Kb + (size_t)b * NN * DD;
  const bf16_t* Hbase = Hb + (size_t)b * CC * NN;

  for (int j0 = 0; j0 < NN; j0 += 64) {
    // prefetch next j-tile (global_prefetch_b8): K rows + this wave's H rows
    __builtin_prefetch(Kbase + (size_t)(j0 + 64 + 16 * ((2 * w) & 3) + n) * DD, 0, 3);
    __builtin_prefetch(Kbase + (size_t)(j0 + 64 + 16 * ((2 * w + 1) & 3) + n) * DD, 0, 3);
    __builtin_prefetch(Hbase + (size_t)(w * 32 + n) * NN + j0 + 64, 0, 3);
    __builtin_prefetch(Hbase + (size_t)(w * 32 + 16 + n) * NN + j0 + 64, 0, 3);

    // ---- S = Q K^T for this 64x64 tile: 16 wmma tiles over 8 waves ----
#pragma unroll
    for (int t = 0; t < 2; ++t) {
      const int tj = (2 * w + t) & 3;
      const bf16_t* krow = Kbase + (size_t)(j0 + 16 * tj + n) * DD + 16 * hi;
      v16bf bk = ld2_bf16(krow, krow + 8);
      v8f s = wmma_bf16(aq, bk, zero_v8f());
#pragma unroll
      for (int r = 0; r < 8; ++r)
        Sld[16 * ti + r + 8 * hi][16 * tj + n] = s[r];
    }
    __syncthreads();

    // ---- online softmax: one thread per query row ----
    if (tid < 64) {
      const float4* rp = (const float4*)&Sld[tid][0];
      float mx = -3.0e38f;
#pragma unroll
      for (int g = 0; g < 16; ++g) {
        float4 f = rp[g];
        mx = fmaxf(mx, fmaxf(fmaxf(f.x, f.y), fmaxf(f.z, f.w)));
      }
      const float mnew = fmaxf(m_run, mx);
      const float sc   = __expf(m_run - mnew);
      float ssum = 0.0f;
      bf16_t* prow = &Pld[tid][0];
#pragma unroll
      for (int g = 0; g < 8; ++g) {
        float4 f0 = rp[2 * g], f1 = rp[2 * g + 1];
        float e0 = __expf(f0.x - mnew), e1 = __expf(f0.y - mnew);
        float e2 = __expf(f0.z - mnew), e3 = __expf(f0.w - mnew);
        float e4 = __expf(f1.x - mnew), e5 = __expf(f1.y - mnew);
        float e6 = __expf(f1.z - mnew), e7 = __expf(f1.w - mnew);
        ssum += (e0 + e1 + e2 + e3) + (e4 + e5 + e6 + e7);
        v8bf pk;
        pk[0] = (bf16_t)e0; pk[1] = (bf16_t)e1; pk[2] = (bf16_t)e2; pk[3] = (bf16_t)e3;
        pk[4] = (bf16_t)e4; pk[5] = (bf16_t)e5; pk[6] = (bf16_t)e6; pk[7] = (bf16_t)e7;
        *(v8bf*)(prow + 8 * g) = pk;
      }
      l_run = l_run * sc + ssum;
      m_run = mnew;
      rs[tid] = sc;
    }
    __syncthreads();

    // ---- rescale accumulators (float4 LDS reads), then O += P * V ----
    const float4* rsv = (const float4*)rs;
#pragma unroll
    for (int ri = 0; ri < 4; ++ri) {
      const int rbase = 16 * ri + 8 * hi;            // multiple of 4
      const float4 s0 = rsv[(rbase >> 2) + 0];
      const float4 s1 = rsv[(rbase >> 2) + 1];
      const float sv[8] = {s0.x, s0.y, s0.z, s0.w, s1.x, s1.y, s1.z, s1.w};
#pragma unroll
      for (int r = 0; r < 8; ++r) {
        acc[ri][0][r] *= sv[r];
        acc[ri][1][r] *= sv[r];
      }
    }
#pragma unroll
    for (int ks = 0; ks < 2; ++ks) {
      v16bf bv[2];
#pragma unroll
      for (int ci = 0; ci < 2; ++ci) {
        const int c = w * 32 + ci * 16 + n;
        const bf16_t* hrow = Hbase + (size_t)c * NN + j0 + 32 * ks + 16 * hi;
        bv[ci] = ld2_bf16(hrow, hrow + 8);
      }
#pragma unroll
      for (int ri = 0; ri < 4; ++ri) {
        const bf16_t* prow = &Pld[16 * ri + n][32 * ks + 16 * hi];
        v16bf ap = ld2_bf16(prow, prow + 8);
        acc[ri][0] = wmma_bf16(ap, bv[0], acc[ri][0]);
        acc[ri][1] = wmma_bf16(ap, bv[1], acc[ri][1]);
      }
    }
  }

  if (tid < 64) ls[tid] = l_run;
  __syncthreads();

  // ---- epilogue: out[b][c][i] = gamma * O[i][c] / l[i] + x[b][c][i] ----
  // accumulator rows r=0..7 map to consecutive i -> b128 loads/stores
  const float gamma = gamma_p[0];
  const float4* lsv = (const float4*)ls;
#pragma unroll
  for (int ri = 0; ri < 4; ++ri) {
    const int rbase = 16 * ri + 8 * hi;
    const float4 l0 = lsv[(rbase >> 2) + 0];
    const float4 l1 = lsv[(rbase >> 2) + 1];
    const float inv[8] = {1.0f / l0.x, 1.0f / l0.y, 1.0f / l0.z, 1.0f / l0.w,
                          1.0f / l1.x, 1.0f / l1.y, 1.0f / l1.z, 1.0f / l1.w};
#pragma unroll
    for (int ci = 0; ci < 2; ++ci) {
      const int c = w * 32 + ci * 16 + n;
      const size_t idx = ((size_t)b * CC + c) * NN + i0 + rbase;
      const float4 x0 = *(const float4*)(x + idx);
      const float4 x1 = *(const float4*)(x + idx + 4);
      float4 o0, o1;
      o0.x = gamma * acc[ri][ci][0] * inv[0] + x0.x;
      o0.y = gamma * acc[ri][ci][1] * inv[1] + x0.y;
      o0.z = gamma * acc[ri][ci][2] * inv[2] + x0.z;
      o0.w = gamma * acc[ri][ci][3] * inv[3] + x0.w;
      o1.x = gamma * acc[ri][ci][4] * inv[4] + x1.x;
      o1.y = gamma * acc[ri][ci][5] * inv[5] + x1.y;
      o1.z = gamma * acc[ri][ci][6] * inv[6] + x1.z;
      o1.w = gamma * acc[ri][ci][7] * inv[7] + x1.w;
      *(float4*)(out + idx)     = o0;
      *(float4*)(out + idx + 4) = o1;
    }
  }
}

extern "C" void kernel_launch(void* const* d_in, const int* in_sizes, int n_in,
                              void* d_out, int out_size, void* d_ws, size_t ws_size,
                              hipStream_t stream) {
  (void)in_sizes; (void)n_in; (void)out_size; (void)ws_size;
  const float* x     = (const float*)d_in[0];
  const float* y     = (const float*)d_in[1];
  const float* Wf    = (const float*)d_in[2];
  const float* bfv   = (const float*)d_in[3];
  const float* Wg    = (const float*)d_in[4];
  const float* bg    = (const float*)d_in[5];
  const float* Whm   = (const float*)d_in[6];
  const float* bh    = (const float*)d_in[7];
  const float* gamma = (const float*)d_in[8];
  float* out = (float*)d_out;

  // workspace layout (total ~52 MB)
  char* ws = (char*)d_ws;
  const size_t szXT = (size_t)BB * NN * CC * sizeof(bf16_t);  // 16 MB
  const size_t szQK = (size_t)BB * NN * DD * sizeof(bf16_t);  // 2 MB
  bf16_t* xT = (bf16_t*)ws;                 ws += szXT;
  bf16_t* yT = (bf16_t*)ws;                 ws += szXT;
  bf16_t* Qb = (bf16_t*)ws;                 ws += szQK;
  bf16_t* Kb = (bf16_t*)ws;                 ws += szQK;
  bf16_t* Hb = (bf16_t*)ws;

  dim3 tgrid(NN / 32, CC / 32, BB), tblk(32, 8);
  k_transpose_cvt<<<tgrid, tblk, 0, stream>>>(x, xT);
  k_transpose_cvt<<<tgrid, tblk, 0, stream>>>(y, yT);

  k_qk_proj<<<dim3(NN / 128, BB), 256, 0, stream>>>(xT, Wf, bfv, Qb);
  k_qk_proj<<<dim3(NN / 128, BB), 256, 0, stream>>>(yT, Wg, bg, Kb);
  k_h_proj<<<dim3(NN / 64, CC / 128, BB), 256, 0, stream>>>(xT, Whm, bh, Hb);

  k_attention<<<dim3(NN / 64, BB), 256, 0, stream>>>(Qb, Kb, Hb, x, gamma, out);
}
